// Preconditioner_10625749090435
// MI455X (gfx1250) — compile-verified, matrix operand order checked by reading, not run on hardware
//
#include <hip/hip_runtime.h>

#ifndef __has_builtin
#define __has_builtin(x) 0
#endif

// ---- CDNA5 feature probes (compile-safe) ------------------------------------
#if __has_builtin(__builtin_amdgcn_wmma_f32_16x16x4_f32)
#define HAVE_WMMA_F32X4 1
#else
#define HAVE_WMMA_F32X4 0
#endif
#if !HAVE_WMMA_F32X4 && __has_builtin(__builtin_amdgcn_wmma_f32_16x16x32_f16)
#define HAVE_WMMA_F16 1
#else
#define HAVE_WMMA_F16 0
#endif
#if __has_builtin(__builtin_amdgcn_global_load_async_to_lds_b32) && \
    __has_builtin(__builtin_amdgcn_s_wait_asynccnt)
#define HAVE_ASYNC_LDS 1
#else
#define HAVE_ASYNC_LDS 0
#endif

typedef __attribute__((ext_vector_type(2)))  float    v2f;
typedef __attribute__((ext_vector_type(8)))  float    v8f;
typedef __attribute__((ext_vector_type(16))) _Float16 v16h;

// ---- Problem constants ------------------------------------------------------
constexpr int   Vn     = 150000;
constexpr int   Fn     = 300000;
constexpr int   EMAX   = 6 * Fn;          // max unique directed edges
constexpr int   HASHSZ = 1 << 22;         // 4M slots, load factor ~0.43
constexpr float SMOOTH = 19.0f;
constexpr float EPS    = 1e-30f;
constexpr int   ITERS  = 24;
constexpr int   TPB    = 256;
constexpr int   NBLK   = (Vn + TPB - 1) / TPB;   // 586
constexpr int   SMEM_COLS = 6144;         // 24KB LDS stage for block's CSR cols
constexpr unsigned long long EMPTYK = ~0ull;

// ---- Exact f32 wave(32) sum using the matrix unit ---------------------------
// A(16x4 f32): lanes 0-15 -> {K0,K1}, lanes 16-31 -> {K2,K3}. With a.x=v,a.y=0
// and B=ones: D(m,n) = v[m] + v[m+16] for every n. Sum the 8 D rows a lane
// holds, then fold the two lane halves with one shfl_xor(16). Exact f32.
__device__ __forceinline__ float wave_sum_f32(float v) {
#if HAVE_WMMA_F32X4
  v2f a; a.x = v;    a.y = 0.0f;
  v2f b; b.x = 1.0f; b.y = 1.0f;
  v8f c = {};
  v8f d = __builtin_amdgcn_wmma_f32_16x16x4_f32(false, a, false, b,
                                                (short)0, c, false, false);
  float s = d[0] + d[1] + d[2] + d[3] + d[4] + d[5] + d[6] + d[7];
  s += __shfl_xor(s, 16, 32);
  return s;
#elif HAVE_WMMA_F16
  v16h a = (v16h)(_Float16)0; a[0] = (_Float16)v;   // A(m,0)=v[m], A(m,8)=v[m+16]
  v16h b;
  #pragma unroll
  for (int i = 0; i < 16; ++i) b[i] = (_Float16)1.0f;
  v8f c = {};
  v8f d = __builtin_amdgcn_wmma_f32_16x16x32_f16(false, a, false, b,
                                                 (short)0, c, false, false);
  float s = d[0] + d[1] + d[2] + d[3] + d[4] + d[5] + d[6] + d[7];
  s += __shfl_xor(s, 16, 32);
  return s;
#else
  #pragma unroll
  for (int off = 16; off > 0; off >>= 1) v += __shfl_xor(v, off, 32);
  return v;
#endif
}

// Block reduction of 3 channels; thread 0 writes dst[0..2].
__device__ __forceinline__ void block_reduce3(float t0, float t1, float t2,
                                              float* dst) {
  __shared__ float wred[TPB / 32][3];
  const int tid = threadIdx.x;
  t0 = wave_sum_f32(t0);
  t1 = wave_sum_f32(t1);
  t2 = wave_sum_f32(t2);
  if ((tid & 31) == 0) {
    const int w = tid >> 5;
    wred[w][0] = t0; wred[w][1] = t1; wred[w][2] = t2;
  }
  __syncthreads();
  if (tid == 0) {
    float s0 = 0.f, s1 = 0.f, s2 = 0.f;
    #pragma unroll
    for (int w = 0; w < TPB / 32; ++w) {
      s0 += wred[w][0]; s1 += wred[w][1]; s2 += wred[w][2];
    }
    dst[0] = s0; dst[1] = s1; dst[2] = s2;
  }
}

// ---- Setup kernels ----------------------------------------------------------
__global__ void k_init(unsigned long long* hashk, unsigned* deg, unsigned* fill,
                       float* scal) {
  const int idx = blockIdx.x * TPB + threadIdx.x;
  hashk[idx] = EMPTYK;                 // grid == HASHSZ threads
  if (idx < Vn) { deg[idx] = 0u; fill[idx] = 0u; }
  if (idx < 192) scal[idx] = 0.0f;     // rs[0..24][3] + pAp[0..23][3]
}

__device__ __forceinline__ unsigned hash_mix(unsigned long long k) {
  k ^= k >> 33; k *= 0xff51afd7ed558ccdULL;
  k ^= k >> 29; k *= 0xc4ceb9fe1a85ec53ULL;
  k ^= k >> 32;
  return (unsigned)k & (HASHSZ - 1);
}

// Dedup the 6F directed edges into an open-addressing hash; count degrees.
__global__ void k_edges(const int* __restrict__ faces,
                        unsigned long long* __restrict__ hashk,
                        unsigned* __restrict__ deg) {
  const long long e = (long long)blockIdx.x * TPB + threadIdx.x;
  if (e >= (long long)EMAX) return;
  const int dir = (e >= (long long)3 * Fn) ? 1 : 0;
  const long long t = dir ? e - (long long)3 * Fn : e;
  const int f = (int)(t / 3), s = (int)(t % 3);
  const unsigned a = (unsigned)faces[f * 3 + ((s + 1) % 3)];   // ii
  const unsigned b = (unsigned)faces[f * 3 + ((s + 2) % 3)];   // jj
  const unsigned src = dir ? b : a;
  const unsigned dst = dir ? a : b;
  const unsigned long long key = ((unsigned long long)src << 32) | dst;
  unsigned idx = hash_mix(key);
  for (;;) {
    const unsigned long long cur = hashk[idx];
    if (cur == key) break;                                   // duplicate
    if (cur == EMPTYK) {
      const unsigned long long old = atomicCAS(&hashk[idx], EMPTYK, key);
      if (old == EMPTYK) { atomicAdd(&deg[src], 1u); break; } // inserted
      if (old == key) break;                                  // dup raced in
    }
    idx = (idx + 1) & (HASHSZ - 1);
  }
}

// Block-local exclusive scan of degrees -> rowp, per-block sums -> bsum.
__global__ void k_scan_a(const unsigned* __restrict__ deg,
                         unsigned* __restrict__ rowp,
                         unsigned* __restrict__ bsum) {
  __shared__ unsigned sc[TPB];
  const int tid = threadIdx.x;
  const int i = blockIdx.x * TPB + tid;
  const unsigned v = (i < Vn) ? deg[i] : 0u;
  sc[tid] = v;
  __syncthreads();
  for (int off = 1; off < TPB; off <<= 1) {
    const unsigned t = (tid >= off) ? sc[tid - off] : 0u;
    __syncthreads();
    sc[tid] += t;
    __syncthreads();
  }
  if (i < Vn) rowp[i] = sc[tid] - v;
  if (tid == TPB - 1) bsum[blockIdx.x] = sc[tid];
}

__global__ void k_scan_b(unsigned* bsum, unsigned* rowp, int n) {
  if (blockIdx.x == 0 && threadIdx.x == 0) {
    unsigned acc = 0;
    for (int i = 0; i < n; ++i) { const unsigned t = bsum[i]; bsum[i] = acc; acc += t; }
    rowp[Vn] = acc;
  }
}

__global__ void k_scan_c(unsigned* __restrict__ rowp,
                         const unsigned* __restrict__ bsum) {
  const int i = blockIdx.x * TPB + threadIdx.x;
  if (i < Vn) rowp[i] += bsum[blockIdx.x];
}

// Sweep the hash table, scatter column indices into CSR.
__global__ void k_scatter(const unsigned long long* __restrict__ hashk,
                          const unsigned* __restrict__ rowp,
                          unsigned* __restrict__ fill,
                          unsigned* __restrict__ ccol) {
  const int idx = blockIdx.x * TPB + threadIdx.x;
  const unsigned long long key = hashk[idx];
  if (key == EMPTYK) return;
  const unsigned src = (unsigned)(key >> 32);
  const unsigned dst = (unsigned)key;
  const unsigned pos = rowp[src] + atomicAdd(&fill[src], 1u);
  ccol[pos] = dst;
}

// ---- CG kernels -------------------------------------------------------------
// M@v row i = v[i] + S*(deg[i]*v[i] - sum_{j in N(i)} v[j]).
// mode 0: invec=verts -> r=p=u, x=0, partial=u*u   (precondition + CG init)
// mode 1: invec=p     -> Ap,               partial=p*Ap
__global__ void k_spmv(int mode, const float* __restrict__ invec,
                       const unsigned* __restrict__ rowp,
                       const unsigned* __restrict__ ccol,
                       float* __restrict__ Ap, float* __restrict__ rv,
                       float* __restrict__ pv, float* __restrict__ xv,
                       float* __restrict__ parts) {
  __shared__ unsigned lds_cols[SMEM_COLS];
  const int tid = threadIdx.x;
  const int r0 = blockIdx.x * TPB;
  const int row = r0 + tid;
  const int rhi = (r0 + TPB < Vn) ? (r0 + TPB) : Vn;
  const unsigned cs = rowp[r0];
  const unsigned ce = rowp[rhi];
  const unsigned len = ce - cs;
  const bool use_lds = (len <= (unsigned)SMEM_COLS);   // block-uniform
  if (use_lds) {
    for (unsigned k = tid; k < len; k += TPB) {
#if HAVE_ASYNC_LDS
      __builtin_amdgcn_global_load_async_to_lds_b32(
          (__attribute__((address_space(1))) int*)(void*)(ccol + cs + k),
          (__attribute__((address_space(3))) int*)(void*)&lds_cols[k], 0, 0);
#else
      lds_cols[k] = ccol[cs + k];
#endif
    }
#if HAVE_ASYNC_LDS
    __builtin_amdgcn_s_wait_asynccnt(0);
#endif
  }
  __syncthreads();

  float t0 = 0.f, t1 = 0.f, t2 = 0.f;
  if (row < Vn) {
    const unsigned rb = rowp[row], re = rowp[row + 1];
    float s0 = 0.f, s1 = 0.f, s2 = 0.f;
    if (use_lds) {
      for (unsigned k = rb; k < re; ++k) {
        const unsigned c = lds_cols[k - cs];
        s0 += invec[c * 3 + 0]; s1 += invec[c * 3 + 1]; s2 += invec[c * 3 + 2];
      }
    } else {
      for (unsigned k = rb; k < re; ++k) {
        const unsigned c = ccol[k];
        s0 += invec[c * 3 + 0]; s1 += invec[c * 3 + 1]; s2 += invec[c * 3 + 2];
      }
    }
    const float dgr = (float)(re - rb);
    const float v0 = invec[row * 3 + 0];
    const float v1 = invec[row * 3 + 1];
    const float v2 = invec[row * 3 + 2];
    const float a0 = v0 + SMOOTH * (dgr * v0 - s0);
    const float a1 = v1 + SMOOTH * (dgr * v1 - s1);
    const float a2 = v2 + SMOOTH * (dgr * v2 - s2);
    if (mode == 0) {
      rv[row * 3 + 0] = a0; rv[row * 3 + 1] = a1; rv[row * 3 + 2] = a2;
      pv[row * 3 + 0] = a0; pv[row * 3 + 1] = a1; pv[row * 3 + 2] = a2;
      xv[row * 3 + 0] = 0.f; xv[row * 3 + 1] = 0.f; xv[row * 3 + 2] = 0.f;
      t0 = a0 * a0; t1 = a1 * a1; t2 = a2 * a2;        // rs0 = |u|^2
    } else {
      Ap[row * 3 + 0] = a0; Ap[row * 3 + 1] = a1; Ap[row * 3 + 2] = a2;
      t0 = v0 * a0; t1 = v1 * a1; t2 = v2 * a2;        // p.Ap
    }
  }
  block_reduce3(t0, t1, t2, parts + (size_t)blockIdx.x * 3);
}

// Deterministic combine of per-block partials -> 3 scalars.
__global__ void k_reduce(const float* __restrict__ parts, int n,
                         float* __restrict__ out3) {
  const int tid = threadIdx.x;
  float s0 = 0.f, s1 = 0.f, s2 = 0.f;
  for (int k = tid; k < n; k += TPB) {
    s0 += parts[k * 3 + 0]; s1 += parts[k * 3 + 1]; s2 += parts[k * 3 + 2];
  }
  block_reduce3(s0, s1, s2, out3);
}

// x += alpha*p ; r -= alpha*Ap ; partial = r*r
__global__ void k_update1(const float* __restrict__ rs_old,
                          const float* __restrict__ pap,
                          const float* __restrict__ pv,
                          const float* __restrict__ Ap,
                          float* __restrict__ xv, float* __restrict__ rv,
                          float* __restrict__ parts) {
  const int row = blockIdx.x * TPB + threadIdx.x;
  const float al0 = rs_old[0] / (pap[0] + EPS);
  const float al1 = rs_old[1] / (pap[1] + EPS);
  const float al2 = rs_old[2] / (pap[2] + EPS);
  float t0 = 0.f, t1 = 0.f, t2 = 0.f;
  if (row < Vn) {
    xv[row * 3 + 0] += al0 * pv[row * 3 + 0];
    xv[row * 3 + 1] += al1 * pv[row * 3 + 1];
    xv[row * 3 + 2] += al2 * pv[row * 3 + 2];
    const float r0n = rv[row * 3 + 0] - al0 * Ap[row * 3 + 0];
    const float r1n = rv[row * 3 + 1] - al1 * Ap[row * 3 + 1];
    const float r2n = rv[row * 3 + 2] - al2 * Ap[row * 3 + 2];
    rv[row * 3 + 0] = r0n; rv[row * 3 + 1] = r1n; rv[row * 3 + 2] = r2n;
    t0 = r0n * r0n; t1 = r1n * r1n; t2 = r2n * r2n;
  }
  block_reduce3(t0, t1, t2, parts + (size_t)blockIdx.x * 3);
}

// p = r + beta*p
__global__ void k_update2(const float* __restrict__ rs_new,
                          const float* __restrict__ rs_old,
                          const float* __restrict__ rv,
                          float* __restrict__ pv) {
  const int row = blockIdx.x * TPB + threadIdx.x;
  if (row >= Vn) return;
  const float b0 = rs_new[0] / (rs_old[0] + EPS);
  const float b1 = rs_new[1] / (rs_old[1] + EPS);
  const float b2 = rs_new[2] / (rs_old[2] + EPS);
  pv[row * 3 + 0] = rv[row * 3 + 0] + b0 * pv[row * 3 + 0];
  pv[row * 3 + 1] = rv[row * 3 + 1] + b1 * pv[row * 3 + 1];
  pv[row * 3 + 2] = rv[row * 3 + 2] + b2 * pv[row * 3 + 2];
}

__global__ void k_faces(const int* __restrict__ faces, float* __restrict__ outF) {
  const int i = blockIdx.x * TPB + threadIdx.x;
  if (i < Fn * 3) outF[i] = (float)faces[i];
}

// ---- Host launcher ----------------------------------------------------------
extern "C" void kernel_launch(void* const* d_in, const int* in_sizes, int n_in,
                              void* d_out, int out_size, void* d_ws, size_t ws_size,
                              hipStream_t stream) {
  (void)in_sizes; (void)n_in; (void)out_size; (void)ws_size;
  const float* verts = (const float*)d_in[0];
  const int*   faces = (const int*)d_in[1];
  float* out = (float*)d_out;

  char* ws = (char*)d_ws;
  size_t off = 0;
  auto alloc = [&](size_t bytes) {
    size_t o = off;
    off = (off + bytes + 255) & ~(size_t)255;
    return o;
  };
  unsigned long long* hashk = (unsigned long long*)(ws + alloc((size_t)HASHSZ * 8)); // 32 MB
  unsigned* deg   = (unsigned*)(ws + alloc((size_t)Vn * 4));
  unsigned* fill  = (unsigned*)(ws + alloc((size_t)Vn * 4));
  unsigned* rowp  = (unsigned*)(ws + alloc((size_t)(Vn + 1) * 4));
  unsigned* ccol  = (unsigned*)(ws + alloc((size_t)EMAX * 4));                        // 7.2 MB
  unsigned* bsum  = (unsigned*)(ws + alloc((size_t)4096 * 4));
  float*    rv    = (float*)(ws + alloc((size_t)Vn * 3 * 4));
  float*    pv    = (float*)(ws + alloc((size_t)Vn * 3 * 4));
  float*    apv   = (float*)(ws + alloc((size_t)Vn * 3 * 4));
  float*    scal  = (float*)(ws + alloc(1024));    // rs[25][3] @0, pAp[24][3] @75
  float*    parts = (float*)(ws + alloc((size_t)NBLK * 3 * 4));
  float*    xv    = out;              // x lives directly in d_out
  float*    outF  = out + (size_t)Vn * 3;

  // ---- one-time graph build (re-run every call; deterministic work) ----
  k_init<<<HASHSZ / TPB, TPB, 0, stream>>>(hashk, deg, fill, scal);
  k_edges<<<(EMAX + TPB - 1) / TPB, TPB, 0, stream>>>(faces, hashk, deg);
  k_scan_a<<<NBLK, TPB, 0, stream>>>(deg, rowp, bsum);
  k_scan_b<<<1, 1, 0, stream>>>(bsum, rowp, NBLK);
  k_scan_c<<<NBLK, TPB, 0, stream>>>(rowp, bsum);
  k_scatter<<<HASHSZ / TPB, TPB, 0, stream>>>(hashk, rowp, fill, ccol);

  // ---- u = M@verts ; r = p = u ; x = 0 ; rs0 = |u|^2 ----
  k_spmv<<<NBLK, TPB, 0, stream>>>(0, verts, rowp, ccol, apv, rv, pv, xv, parts);
  k_reduce<<<1, TPB, 0, stream>>>(parts, NBLK, scal + 0);

  // ---- 24 CG iterations ----
  for (int it = 0; it < ITERS; ++it) {
    float* rs_old = scal + it * 3;
    float* rs_new = scal + (it + 1) * 3;
    float* pap    = scal + 75 + it * 3;
    k_spmv<<<NBLK, TPB, 0, stream>>>(1, pv, rowp, ccol, apv, rv, pv, xv, parts);
    k_reduce<<<1, TPB, 0, stream>>>(parts, NBLK, pap);
    k_update1<<<NBLK, TPB, 0, stream>>>(rs_old, pap, pv, apv, xv, rv, parts);
    k_reduce<<<1, TPB, 0, stream>>>(parts, NBLK, rs_new);
    k_update2<<<NBLK, TPB, 0, stream>>>(rs_new, rs_old, rv, pv);
  }

  // ---- second tuple element: faces ----
  k_faces<<<(Fn * 3 + TPB - 1) / TPB, TPB, 0, stream>>>(faces, outF);
}
